// TransformerModel_14491219656904
// MI455X (gfx1250) — matrix operand order repaired and checked by reading, not compiled, not run on hardware
//
#include <hip/hip_runtime.h>

// ============================================================================
// CDNA5 (gfx1250) transformer forward.  All GEMMs via V_WMMA_F32_16X16X32_BF16
// (wave32, 16x16x32, f32 accumulate).  fp32 activations/weights are converted
// to bf16 while staging tiles into LDS.  128x64 block tile, 8 waves, each wave
// owns a 32x32 sub-tile (4 WMMAs per k-step) with software-pipelined global
// loads and global_prefetch_b8 lookahead.
// ============================================================================

typedef __attribute__((ext_vector_type(8)))  __bf16 bf16x8;
typedef __attribute__((ext_vector_type(16))) __bf16 bf16x16;
typedef __attribute__((ext_vector_type(8)))  float  f32x8;

namespace {
constexpr int   DD  = 512;     // model dim
constexpr int   HH  = 8;       // heads
constexpr int   DH_ = 64;      // head dim
constexpr int   FF_ = 2048;    // ffn dim
constexpr int   NL_ = 6;       // layers
constexpr int   VV  = 32000;   // vocab
constexpr int   BB  = 8;       // batch
constexpr int   LS_ = 256;     // src len
constexpr int   LT_ = 256;     // trg len
constexpr int   PP  = 301;     // rel-pos table rows (2*150+1)
constexpr int   PLD = 304;     // padded leading dim for T
constexpr float EMB_SCALE_ = 22.627416997969522f;  // sqrt(512)
constexpr float INV_SCALE_ = 0.125f;               // 1/sqrt(64)
}

// ----------------------------------------------------------------------------
// Build a 16-element bf16 fragment from two 16-byte LDS chunks.
// ----------------------------------------------------------------------------
__device__ inline bf16x16 make_frag(const __bf16* p0, const __bf16* p1) {
  bf16x8 lo = *(const bf16x8*)p0;
  bf16x8 hi = *(const bf16x8*)p1;
  bf16x16 r;
#pragma unroll
  for (int i = 0; i < 8; ++i) { r[i] = lo[i]; r[i + 8] = hi[i]; }
  return r;
}

// ----------------------------------------------------------------------------
// Batched GEMM:  C = act(alpha * A @ B (+bias))
//   A: [M,K] row-major (row stride lda), batch ptr = A + outer*sAo + inner*sAi
//   B: TRANSB ? B given as [N,K] (i.e. we compute A @ B^T)
//             : B given as [K,N] row-major
//   C: [M,N] row-major (row stride ldc)
// batch index bz: outer = bz / innerB, inner = bz % innerB (lets one kernel
// address [B][L][H*DH] head slices: outer=b stride, inner=h stride).
// Block: 256 threads = 8 waves; tile 128(M) x 64(N), BK=32.
// Wave (wm=w&3, wn=w>>2) owns a 32x32 sub-tile -> 4 WMMA accumulators.
// Global loads for step k+1 are issued while WMMAs for step k execute.
// ----------------------------------------------------------------------------
template <bool TRANSB>
__global__ __launch_bounds__(256) void wmma_gemm_kernel(
    const float* __restrict__ A, int lda, long long sAo, long long sAi,
    const float* __restrict__ Bm, int ldb, long long sBo, long long sBi,
    float* __restrict__ C, int ldc, long long sCo, long long sCi,
    int M, int N, int K, int innerB,
    const float* __restrict__ bias, float alpha, int relu) {
  __shared__ __align__(16) __bf16 sA[128 * 32];  // [m][k]
  __shared__ __align__(16) __bf16 sB[64 * 32];   // [n][k]

  const int bz = blockIdx.z;
  const int outer = bz / innerB, inner = bz % innerB;
  const float* Ab = A + (long long)outer * sAo + (long long)inner * sAi;
  const float* Bb = Bm + (long long)outer * sBo + (long long)inner * sBi;
  float* Cb = C + (long long)outer * sCo + (long long)inner * sCi;

  const int m0 = blockIdx.y * 128;
  const int n0 = blockIdx.x * 64;
  const int t = threadIdx.x;
  const int lane = t & 31;
  const int wave = t >> 5;
  const int wm = wave & 3;   // 4 wave-rows of 32
  const int wn = wave >> 2;  // 2 wave-cols of 32

  f32x8 acc00, acc01, acc10, acc11;
#pragma unroll
  for (int i = 0; i < 8; ++i) {
    acc00[i] = 0.f; acc01[i] = 0.f; acc10[i] = 0.f; acc11[i] = 0.f;
  }

  const int lr = t >> 2;        // 0..63 tile row (A rows lr, lr+64; B row lr)
  const int lc = (t & 3) * 8;   // 0,8,16,24 k-chunk

  const int gmA0 = m0 + lr;
  const int gmA1 = m0 + lr + 64;
  const int gnB  = n0 + lr;

  float a0[8], a1[8], b0[8];

  // ---- staged global loads for k-step k0 ----
  auto load_tiles = [&](int k0) {
#pragma unroll
    for (int i = 0; i < 8; ++i) { a0[i] = 0.f; a1[i] = 0.f; b0[i] = 0.f; }
    if (gmA0 < M) {
      const float* s = Ab + (long long)gmA0 * lda + (k0 + lc);
#pragma unroll
      for (int i = 0; i < 8; ++i) a0[i] = s[i];
    }
    if (gmA1 < M) {
      const float* s = Ab + (long long)gmA1 * lda + (k0 + lc);
#pragma unroll
      for (int i = 0; i < 8; ++i) a1[i] = s[i];
    }
    if (gnB < N) {
      if (TRANSB) {
        const float* s = Bb + (long long)gnB * ldb + (k0 + lc);
#pragma unroll
        for (int i = 0; i < 8; ++i) b0[i] = s[i];
      } else {
        const float* s = Bb + (long long)(k0 + lc) * ldb + gnB;
#pragma unroll
        for (int i = 0; i < 8; ++i) b0[i] = s[(long long)i * ldb];
      }
    }
  };

  load_tiles(0);

  for (int k0 = 0; k0 < K; k0 += 32) {
    // ---- commit staged registers to LDS (fp32 -> bf16) ----
#pragma unroll
    for (int i = 0; i < 8; ++i) sA[lr * 32 + lc + i]        = (__bf16)a0[i];
#pragma unroll
    for (int i = 0; i < 8; ++i) sA[(lr + 64) * 32 + lc + i] = (__bf16)a1[i];
#pragma unroll
    for (int i = 0; i < 8; ++i) sB[lr * 32 + lc + i]        = (__bf16)b0[i];
    __syncthreads();

    // ---- software pipeline: issue next tile's loads + L2 prefetch ----
    if (k0 + 32 < K) {
      load_tiles(k0 + 32);
      if (k0 + 64 < K) {
        if (gmA0 < M) __builtin_prefetch(Ab + (long long)gmA0 * lda + k0 + 64, 0, 3);
        if (gnB < N) {
          if (TRANSB) __builtin_prefetch(Bb + (long long)gnB * ldb + k0 + 64, 0, 3);
          else        __builtin_prefetch(Bb + (long long)(k0 + 64) * ldb + gnB, 0, 3);
        }
      }
    }

    // ---- fragments per ISA layout ----
    // A 16x32 bf16: lane row = lane&15, kbase = (lane>=16)?8:0,
    //               chunks [kbase,kbase+8) and [kbase+16,kbase+24)
    const int ra0 = wm * 32 + (lane & 15);
    const int kbA = (lane >> 4) * 8;
    bf16x16 af0 = make_frag(sA + ra0 * 32 + kbA, sA + ra0 * 32 + kbA + 16);
    bf16x16 af1 = make_frag(sA + (ra0 + 16) * 32 + kbA, sA + (ra0 + 16) * 32 + kbA + 16);

    // B 32x16 bf16 (stored [n][k]): lane col = lane&15, kbase = (lane>=16)?8:0,
    //               chunks [kbase,kbase+8) and [kbase+16,kbase+24)
    const int rb0 = wn * 32 + (lane & 15);
    const int kbB = (lane >> 4) * 8;
    bf16x16 bf0 = make_frag(sB + rb0 * 32 + kbB, sB + rb0 * 32 + kbB + 16);
    bf16x16 bf1 = make_frag(sB + (rb0 + 16) * 32 + kbB, sB + (rb0 + 16) * 32 + kbB + 16);

    acc00 = __builtin_amdgcn_wmma_f32_16x16x32_bf16(false, af0, false, bf0,
                                                    (short)0, acc00, false, false);
    acc01 = __builtin_amdgcn_wmma_f32_16x16x32_bf16(false, af0, false, bf1,
                                                    (short)0, acc01, false, false);
    acc10 = __builtin_amdgcn_wmma_f32_16x16x32_bf16(false, af1, false, bf0,
                                                    (short)0, acc10, false, false);
    acc11 = __builtin_amdgcn_wmma_f32_16x16x32_bf16(false, af1, false, bf1,
                                                    (short)0, acc11, false, false);
    __syncthreads();
  }

  // ---- store C (f32 C/D layout: n = lane&15, m = r + 8*(lane>=16)) ----
  const int nl = lane & 15;
  const int mo = (lane >> 4) * 8;

  auto store_frag = [&](const f32x8& acc, int fm, int fn) {
    const int gn = n0 + wn * 32 + fn * 16 + nl;
    if (gn >= N) return;
    const float bv = bias ? bias[gn] : 0.f;
#pragma unroll
    for (int r = 0; r < 8; ++r) {
      const int gm = m0 + wm * 32 + fm * 16 + mo + r;
      if (gm < M) {
        float v = acc[r] * alpha + bv;
        if (relu) v = fmaxf(v, 0.f);
        Cb[(long long)gm * ldc + gn] = v;
      }
    }
  };
  store_frag(acc00, 0, 0);
  store_frag(acc01, 0, 1);
  store_frag(acc10, 1, 0);
  store_frag(acc11, 1, 1);
}

// ----------------------------------------------------------------------------
// token embedding * sqrt(D).  grid = rows, block = 256, D = 512.
// ----------------------------------------------------------------------------
__global__ __launch_bounds__(256) void embed_kernel(
    const int* __restrict__ tok, const float* __restrict__ tab,
    float* __restrict__ out, float scale) {
  const int row = blockIdx.x;
  const int t = threadIdx.x;
  const long long ob = (long long)row * DD;
  const long long tb = (long long)tok[row] * DD;
  out[ob + t]       = tab[tb + t] * scale;
  out[ob + t + 256] = tab[tb + t + 256] * scale;
}

// ----------------------------------------------------------------------------
// out = LayerNorm(a (+ res)) * g + b.  One block per row (D = 512).
// ----------------------------------------------------------------------------
__global__ __launch_bounds__(256) void add_ln_kernel(
    const float* __restrict__ a, const float* __restrict__ res,
    const float* __restrict__ g, const float* __restrict__ bb,
    float* __restrict__ out) {
  const int row = blockIdx.x;
  const int t = threadIdx.x;
  const long long base = (long long)row * DD;
  float v0 = a[base + t], v1 = a[base + t + 256];
  if (res) { v0 += res[base + t]; v1 += res[base + t + 256]; }

  __shared__ float red[256];
  __shared__ float stat;
  red[t] = v0 + v1;
  __syncthreads();
  for (int s = 128; s > 0; s >>= 1) {
    if (t < s) red[t] += red[t + s];
    __syncthreads();
  }
  if (t == 0) stat = red[0] * (1.f / DD);
  __syncthreads();
  const float mean = stat;
  const float d0 = v0 - mean, d1 = v1 - mean;
  __syncthreads();
  red[t] = d0 * d0 + d1 * d1;
  __syncthreads();
  for (int s = 128; s > 0; s >>= 1) {
    if (t < s) red[t] += red[t + s];
    __syncthreads();
  }
  if (t == 0) stat = red[0] * (1.f / DD);
  __syncthreads();
  const float rstd = rsqrtf(stat + 1e-5f);
  out[base + t]       = d0 * rstd * g[t] + bb[t];
  out[base + t + 256] = d1 * rstd * g[t + 256] + bb[t + 256];
}

// ----------------------------------------------------------------------------
// In-place softmax over key dim with relative-position add + masking.
// S[bh, i, :] already holds (QK^T)/sqrt(d); T[bh, i, r] holds (q.rp[r])/sqrt(d).
// r = clip(j-i, -150, 150) + 149, with -1 wrapping to 300 (torch semantics).
// grid = (Lq, B*H), block = 256 (== Lk).
// ----------------------------------------------------------------------------
__global__ __launch_bounds__(256) void softmax_rel_kernel(
    float* __restrict__ S, const float* __restrict__ T,
    const unsigned char* __restrict__ pad, const unsigned char* __restrict__ peek,
    int Lq, int Lk, int H_) {
  const int i = blockIdx.x;
  const int bh = blockIdx.y;
  const int b = bh / H_;
  const int t = threadIdx.x;
  const long long base = ((long long)bh * Lq + i) * Lk;

  float sval = -INFINITY;
  if (t < Lk) {
    sval = S[base + t];
    int dj = t - i;
    dj = min(150, max(-150, dj));
    int r = dj + 149;
    if (r < 0) r += PP;  // index -1 wraps to 300
    sval += T[((long long)bh * Lq + i) * PLD + r];
    bool m = (pad[(long long)b * Lk + t] != 0);
    if (peek) m = m || (peek[(long long)i * Lk + t] != 0);
    if (m) sval = -INFINITY;
  }

  __shared__ float red[256];
  red[t] = sval;
  __syncthreads();
  for (int s = 128; s > 0; s >>= 1) {
    if (t < s) red[t] = fmaxf(red[t], red[t + s]);
    __syncthreads();
  }
  const float mx = red[0];
  __syncthreads();
  const float e = (t < Lk && mx != -INFINITY) ? expf(sval - mx) : 0.f;
  red[t] = e;
  __syncthreads();
  for (int s = 128; s > 0; s >>= 1) {
    if (t < s) red[t] += red[t + s];
    __syncthreads();
  }
  const float sum = red[0];
  const float inv = (sum > 0.f) ? 1.f / sum : 0.f;
  if (t < Lk) S[base + t] = e * inv;
}

// ============================================================================
// Host-side orchestration
// ============================================================================
namespace {

struct Attn {
  const float *wq, *bq, *wk, *bk, *wv, *bv, *wo, *bo, *rp;
};

inline void launch_gemm(hipStream_t st, bool transB,
                        const float* A, int lda, long long sAo, long long sAi,
                        const float* Bm, int ldb, long long sBo, long long sBi,
                        float* C, int ldc, long long sCo, long long sCi,
                        int M, int N, int K, int batch, int innerB,
                        const float* bias, float alpha, int relu) {
  dim3 grid((N + 63) / 64, (M + 127) / 128, batch);
  if (transB)
    wmma_gemm_kernel<true><<<grid, 256, 0, st>>>(A, lda, sAo, sAi, Bm, ldb, sBo,
                                                 sBi, C, ldc, sCo, sCi, M, N, K,
                                                 innerB, bias, alpha, relu);
  else
    wmma_gemm_kernel<false><<<grid, 256, 0, st>>>(A, lda, sAo, sAi, Bm, ldb, sBo,
                                                  sBi, C, ldc, sCo, sCi, M, N, K,
                                                  innerB, bias, alpha, relu);
}

// Relative-position multi-head attention: out[Mq, D]
void run_mha(hipStream_t st, const Attn& p, const float* xq, const float* xkv,
             int Lq, int Lk, const unsigned char* pad, const unsigned char* peek,
             float* Qb, float* Kb, float* Vb, float* Tb, float* Sb, float* ctx,
             float* out) {
  const int Mq = BB * Lq, Mk = BB * Lk;
  // Q/K/V projections
  launch_gemm(st, false, xq, DD, 0, 0, p.wq, DD, 0, 0, Qb, DD, 0, 0,
              Mq, DD, DD, 1, 1, p.bq, 1.f, 0);
  launch_gemm(st, false, xkv, DD, 0, 0, p.wk, DD, 0, 0, Kb, DD, 0, 0,
              Mk, DD, DD, 1, 1, p.bk, 1.f, 0);
  launch_gemm(st, false, xkv, DD, 0, 0, p.wv, DD, 0, 0, Vb, DD, 0, 0,
              Mk, DD, DD, 1, 1, p.bv, 1.f, 0);
  // T[bh] = (Q_head @ rp^T) / sqrt(d)   [Lq x 301]
  launch_gemm(st, true, Qb, DD, (long long)Lq * DD, DH_, p.rp, DH_, 0, 0,
              Tb, PLD, (long long)HH * Lq * PLD, (long long)Lq * PLD,
              Lq, PP, DH_, BB * HH, HH, nullptr, INV_SCALE_, 0);
  // S[bh] = (Q_head @ K_head^T) / sqrt(d)   [Lq x Lk]
  launch_gemm(st, true, Qb, DD, (long long)Lq * DD, DH_,
              Kb, DD, (long long)Lk * DD, DH_,
              Sb, Lk, (long long)HH * Lq * Lk, (long long)Lq * Lk,
              Lq, Lk, DH_, BB * HH, HH, nullptr, INV_SCALE_, 0);
  // softmax with rel-pos gather + masking (in place)
  softmax_rel_kernel<<<dim3(Lq, BB * HH), 256, 0, st>>>(Sb, Tb, pad, peek, Lq,
                                                        Lk, HH);
  // ctx_head = P @ V_head   (written into [B, Lq, D] at column h*64)
  launch_gemm(st, false, Sb, Lk, (long long)HH * Lq * Lk, (long long)Lq * Lk,
              Vb, DD, (long long)Lk * DD, DH_,
              ctx, DD, (long long)Lq * DD, DH_,
              Lq, DH_, Lk, BB * HH, HH, nullptr, 1.f, 0);
  // output projection
  launch_gemm(st, false, ctx, DD, 0, 0, p.wo, DD, 0, 0, out, DD, 0, 0,
              Mq, DD, DD, 1, 1, p.bo, 1.f, 0);
}

void run_ffn(hipStream_t st, const float* x, const float* w1, const float* b1,
             const float* w2, const float* b2, float* hbuf, float* out, int M) {
  launch_gemm(st, false, x, DD, 0, 0, w1, FF_, 0, 0, hbuf, FF_, 0, 0,
              M, FF_, DD, 1, 1, b1, 1.f, 1 /*relu*/);
  launch_gemm(st, false, hbuf, FF_, 0, 0, w2, DD, 0, 0, out, DD, 0, 0,
              M, DD, FF_, 1, 1, b2, 1.f, 0);
}

inline Attn attn_layer(const float* wq, const float* bq, const float* wk,
                       const float* bk, const float* wv, const float* bv,
                       const float* wo, const float* bo, const float* rp,
                       int l) {
  Attn a;
  a.wq = wq + (long long)l * DD * DD;  a.bq = bq + (long long)l * DD;
  a.wk = wk + (long long)l * DD * DD;  a.bk = bk + (long long)l * DD;
  a.wv = wv + (long long)l * DD * DD;  a.bv = bv + (long long)l * DD;
  a.wo = wo + (long long)l * DD * DD;  a.bo = bo + (long long)l * DD;
  a.rp = rp + (long long)l * PP * DH_;
  return a;
}

}  // namespace

// ----------------------------------------------------------------------------
// Input order: setup_inputs() dict order, nested pytrees flattened in jax
// sorted-key order:
//  0 src  1 trg  2 src_pad_mask  3 trg_pad_mask  4 peek_mask
//  5 src_emb  6 trg_emb  7 gen_w  8 gen_b
//  enc_params: 9 fnb 10 fng
//    layers (sorted): 11 attn.bk 12 attn.bo 13 attn.bq 14 attn.bv 15 attn.rp
//                     16 attn.wk 17 attn.wo 18 attn.wq 19 attn.wv
//                     20 b1 21 b2 22 ln1b 23 ln1g 24 ln2b 25 ln2g 26 w1 27 w2
//  dec_params: 28 fnb 29 fng
//    layers (sorted): 30 b1 31 b2
//                     32 cross.bk 33 cross.bo 34 cross.bq 35 cross.bv 36 cross.rp
//                     37 cross.wk 38 cross.wo 39 cross.wq 40 cross.wv
//                     41 ln1b 42 ln1g 43 ln2b 44 ln2g 45 ln3b 46 ln3g
//                     47 self.bk 48 self.bo 49 self.bq 50 self.bv 51 self.rp
//                     52 self.wk 53 self.wo 54 self.wq 55 self.wv
//                     56 w1 57 w2
// ----------------------------------------------------------------------------
extern "C" void kernel_launch(void* const* d_in, const int* in_sizes, int n_in,
                              void* d_out, int out_size, void* d_ws,
                              size_t ws_size, hipStream_t stream) {
  (void)in_sizes; (void)n_in; (void)out_size; (void)ws_size;

  const int* src = (const int*)d_in[0];
  const int* trg = (const int*)d_in[1];
  const unsigned char* spm  = (const unsigned char*)d_in[2];  // [B,1,LS]
  const unsigned char* tpm  = (const unsigned char*)d_in[3];  // [B,1,LT]
  const unsigned char* peek = (const unsigned char*)d_in[4];  // [1,LT,LT]
  const float* src_emb = (const float*)d_in[5];
  const float* trg_emb = (const float*)d_in[6];
  const float* gen_w = (const float*)d_in[7];
  const float* gen_b = (const float*)d_in[8];

  const float* e_fnb = (const float*)d_in[9];
  const float* e_fng = (const float*)d_in[10];
  const float* dec_fnb = (const float*)d_in[28];
  const float* dec_fng = (const float*)d_in[29];

  // ---- workspace layout (floats) ----
  const int NT = BB * LS_;  // 2048 token rows (enc == dec here)
  float* w = (float*)d_ws;
  float* Qb  = w; w += (long long)NT * DD;
  float* Kb  = w; w += (long long)NT * DD;
  float* Vb  = w; w += (long long)NT * DD;
  float* Tb  = w; w += (long long)BB * HH * LT_ * PLD;
  float* Sb  = w; w += (long long)BB * HH * LT_ * LT_;
  float* ctx = w; w += (long long)NT * DD;
  float* xb  = w; w += (long long)NT * DD;   // encoder activations
  float* yb  = w; w += (long long)NT * DD;   // decoder activations
  float* Eb  = w; w += (long long)NT * DD;   // encoder output (post-LN)
  float* tmp = w; w += (long long)NT * DD;
  float* ffn = w; w += (long long)NT * FF_;

  // ================= encoder =================
  embed_kernel<<<NT, 256, 0, stream>>>(src, src_emb, xb, EMB_SCALE_);
  for (int l = 0; l < NL_; ++l) {
    Attn ap = attn_layer((const float*)d_in[18], (const float*)d_in[13],
                         (const float*)d_in[16], (const float*)d_in[11],
                         (const float*)d_in[19], (const float*)d_in[14],
                         (const float*)d_in[17], (const float*)d_in[12],
                         (const float*)d_in[15], l);
    run_mha(stream, ap, xb, xb, LS_, LS_, spm, nullptr,
            Qb, Kb, Vb, Tb, Sb, ctx, tmp);
    add_ln_kernel<<<NT, 256, 0, stream>>>(
        tmp, xb, (const float*)d_in[23] + (long long)l * DD,
        (const float*)d_in[22] + (long long)l * DD, xb);
    run_ffn(stream, xb, (const float*)d_in[26] + (long long)l * DD * FF_,
            (const float*)d_in[20] + (long long)l * FF_,
            (const float*)d_in[27] + (long long)l * FF_ * DD,
            (const float*)d_in[21] + (long long)l * DD, ffn, tmp, NT);
    add_ln_kernel<<<NT, 256, 0, stream>>>(
        tmp, xb, (const float*)d_in[25] + (long long)l * DD,
        (const float*)d_in[24] + (long long)l * DD, xb);
  }
  add_ln_kernel<<<NT, 256, 0, stream>>>(xb, nullptr, e_fng, e_fnb, Eb);

  // ================= decoder =================
  embed_kernel<<<NT, 256, 0, stream>>>(trg, trg_emb, yb, EMB_SCALE_);
  for (int l = 0; l < NL_; ++l) {
    Attn sp = attn_layer((const float*)d_in[54], (const float*)d_in[49],
                         (const float*)d_in[52], (const float*)d_in[47],
                         (const float*)d_in[55], (const float*)d_in[50],
                         (const float*)d_in[53], (const float*)d_in[48],
                         (const float*)d_in[51], l);
    Attn cp = attn_layer((const float*)d_in[39], (const float*)d_in[34],
                         (const float*)d_in[37], (const float*)d_in[32],
                         (const float*)d_in[40], (const float*)d_in[35],
                         (const float*)d_in[38], (const float*)d_in[33],
                         (const float*)d_in[36], l);
    // masked self-attention (pad | causal)
    run_mha(stream, sp, yb, yb, LT_, LT_, tpm, peek,
            Qb, Kb, Vb, Tb, Sb, ctx, tmp);
    add_ln_kernel<<<NT, 256, 0, stream>>>(
        tmp, yb, (const float*)d_in[42] + (long long)l * DD,
        (const float*)d_in[41] + (long long)l * DD, yb);
    // cross attention (keys/values from encoder output)
    run_mha(stream, cp, yb, Eb, LT_, LS_, spm, nullptr,
            Qb, Kb, Vb, Tb, Sb, ctx, tmp);
    add_ln_kernel<<<NT, 256, 0, stream>>>(
        tmp, yb, (const float*)d_in[44] + (long long)l * DD,
        (const float*)d_in[43] + (long long)l * DD, yb);
    run_ffn(stream, yb, (const float*)d_in[56] + (long long)l * DD * FF_,
            (const float*)d_in[30] + (long long)l * FF_,
            (const float*)d_in[57] + (long long)l * FF_ * DD,
            (const float*)d_in[31] + (long long)l * DD, ffn, tmp, NT);
    add_ln_kernel<<<NT, 256, 0, stream>>>(
        tmp, yb, (const float*)d_in[46] + (long long)l * DD,
        (const float*)d_in[45] + (long long)l * DD, yb);
  }
  add_ln_kernel<<<NT, 256, 0, stream>>>(yb, nullptr, dec_fng, dec_fnb, tmp);

  // ================= logits =================
  launch_gemm(stream, false, tmp, DD, 0, 0, gen_w, VV, 0, 0, (float*)d_out, VV,
              0, 0, NT, VV, DD, 1, 1, gen_b, 1.f, 0);
}
